// CausalSelfAttention_38972533244374
// MI455X (gfx1250) — compile-verified, matrix operand order checked by reading, not run on hardware
//
#include <hip/hip_runtime.h>
#include <stdint.h>

// ---------------------------------------------------------------------------
// Causal self-attention for MI455X (gfx1250), bf16 WMMA pipeline.
// B=4, T=2048, C=1024, H=16, hd=64.
// GEMMs: TDM (tensor_load_to_lds) staged 128x128 block tiles, 8 waves of
// 32x64 (2x4 WMMA tiles). Attention: flash-style, fragments straight from
// global (K row-major, V pre-transposed), online softmax via shfl_xor.
// ---------------------------------------------------------------------------

typedef __attribute__((ext_vector_type(16))) __bf16    v16bf;
typedef __attribute__((ext_vector_type(8)))  float     v8f;
typedef __attribute__((ext_vector_type(4)))  unsigned  v4u;
typedef __attribute__((ext_vector_type(8)))  int       v8i;
typedef __attribute__((ext_vector_type(4)))  int       v4i;

#define NHEAD 16
#define HD    64
#define TSEQ  2048
#define BATCH 4
#define CDIM  1024
#define MTOT  (BATCH * TSEQ)   /* 8192 */
#define N3C   (3 * CDIM)       /* 3072 */

// Round-to-nearest-even fp32 -> bf16.
static __device__ __forceinline__ __bf16 f2bf(float f) {
  unsigned u = __builtin_bit_cast(unsigned, f);
  unsigned r = u + 0x7FFFu + ((u >> 16) & 1u);
  unsigned short h = (unsigned short)(r >> 16);
  return __builtin_bit_cast(__bf16, h);
}

static __device__ __forceinline__ v8f wmma_bf16(const v16bf& a, const v16bf& b, const v8f& c) {
  return __builtin_amdgcn_wmma_f32_16x16x32_bf16(false, a, false, b, (short)0, c, false, false);
}

union Frag { v16bf v; uint4 q[2]; };

// ---------------------------------------------------------------------------
// TDM: 2D tile load global -> LDS.
//   tile/tensor: 32 (dim0, K-elems) x 128 (dim1, rows), bf16 (data_size=2B),
//   row stride = ld elements. LDS padding: 4 dwords (8 bf16) after every
//   16 dwords (32 bf16) -> matches [rows][40] padded LDS layout.
// D# bit layout per CDNA5 ISA 8.3/8.4 (group0 128b, group1 256b).
// ---------------------------------------------------------------------------
static __device__ __forceinline__ void tdm_load_tile_2d(
    unsigned lds_byte_off, const void* gaddr, unsigned long long ld_elems) {
  unsigned long long ga = (unsigned long long)(uintptr_t)gaddr;
  v4u g0;
  g0.x = 1u;                                   // count=1, is_restore=0
  g0.y = lds_byte_off;                         // lds_addr (bytes)
  g0.z = (unsigned)ga;                         // global_addr[31:0]
  g0.w = (unsigned)((ga >> 32) & 0x01FFFFFFu)  // global_addr[56:32]
       | (2u << 30);                           // type = 2 ("image")
  const unsigned td0 = 32u, td1 = 128u;        // tensor dims (exact tile fit)
  const unsigned tl0 = 32u, tl1 = 128u;        // tile dims
  unsigned w0 = (1u << 16)                     // data_size = 1 -> 2 bytes
              | (1u << 20)                     // pad_enable
              | (3u << 22)                     // pad_interval: 16 dwords
              | (3u << 25);                    // pad_amount:   4 dwords
  v8i g1;
  g1[0] = (int)w0;                                         // mask=0, flags
  g1[1] = (int)((td0 & 0xFFFFu) << 16);                    // tensor_dim0 lo
  g1[2] = (int)((td0 >> 16) | ((td1 & 0xFFFFu) << 16));    // d0 hi | d1 lo
  g1[3] = (int)((td1 >> 16) | (tl0 << 16));                // d1 hi | tile0
  g1[4] = (int)(tl1 & 0xFFFFu);                            // tile1 | tile2=0
  g1[5] = (int)(unsigned)(ld_elems & 0xFFFFFFFFull);       // dim0_stride lo
  g1[6] = (int)(unsigned)((ld_elems >> 32) & 0xFFFFull);   // stride hi | d1s
  g1[7] = 0;
  v4i z4 = {0, 0, 0, 0};
#if __clang_major__ >= 23
  v8i z8 = {0, 0, 0, 0, 0, 0, 0, 0};
  __builtin_amdgcn_tensor_load_to_lds(g0, g1, z4, z4, z8, 0);
#else
  __builtin_amdgcn_tensor_load_to_lds(g0, g1, z4, z4, 0);
#endif
}

// ---------------------------------------------------------------------------
// Prep kernel 1: elementwise fp32 -> bf16 (vectorized x4).
// ---------------------------------------------------------------------------
__global__ void cvt_bf16_kernel(const float* __restrict__ in,
                                __bf16* __restrict__ out, int n4) {
  int i = blockIdx.x * blockDim.x + threadIdx.x;
  int stride = gridDim.x * blockDim.x;
  for (; i < n4; i += stride) {
    float4 f = ((const float4*)in)[i];
    unsigned s0 = (unsigned)__builtin_bit_cast(unsigned short, f2bf(f.x));
    unsigned s1 = (unsigned)__builtin_bit_cast(unsigned short, f2bf(f.y));
    unsigned s2 = (unsigned)__builtin_bit_cast(unsigned short, f2bf(f.z));
    unsigned s3 = (unsigned)__builtin_bit_cast(unsigned short, f2bf(f.w));
    uint2 p;
    p.x = s0 | (s1 << 16);
    p.y = s2 | (s3 << 16);
    ((uint2*)out)[i] = p;
  }
}

// ---------------------------------------------------------------------------
// Prep kernel 2: transpose + convert.  in: fp32 [K][N] -> out: bf16 [N][K].
// ---------------------------------------------------------------------------
__global__ __launch_bounds__(256)
void transpose_cvt_kernel(const float* __restrict__ in,
                          __bf16* __restrict__ out, int K, int N) {
  __shared__ float tile[32][33];
  int k0 = blockIdx.y * 32;
  int n0 = blockIdx.x * 32;
  for (int r = threadIdx.y; r < 32; r += 8)
    tile[r][threadIdx.x] = in[(size_t)(k0 + r) * N + n0 + threadIdx.x];
  __syncthreads();
  for (int r = threadIdx.y; r < 32; r += 8)
    out[(size_t)(n0 + r) * K + k0 + threadIdx.x] = f2bf(tile[threadIdx.x][r]);
}

// ---------------------------------------------------------------------------
// GEMM: C[M][N] = A[M][K] (bf16, row-major) * Bt[N][K]^T (bf16) + bias[N].
// Block tile 128x128, 8 waves of 32x64 (2x4 WMMA tiles), K-step 32.
// Tile staging: TDM descriptors issued by wave 0, s_wait_tensorcnt + barrier.
// MODE 0: scatter epilogue into Q / K / V^T (bf16).  MODE 1: fp32 output.
// ---------------------------------------------------------------------------
template <int MODE>
__global__ __launch_bounds__(256)
void gemm_bf16_kernel(const __bf16* __restrict__ A,
                      const __bf16* __restrict__ Bt,
                      const float*  __restrict__ bias,
                      __bf16* __restrict__ Qb,
                      __bf16* __restrict__ Kb,
                      __bf16* __restrict__ Vt,
                      float*  __restrict__ out,
                      int M, int N, int K) {
  __shared__ __bf16 As[128][40];   // 32 K-elems + 8 pad per row (TDM-padded)
  __shared__ __bf16 Bs[128][40];

  const int m0 = blockIdx.y * 128;
  const int n0 = blockIdx.x * 128;
  const int t  = threadIdx.x;
  const int wave = t >> 5;
  const int lane = t & 31;
  const int mw = wave >> 1;        // 0..3 : 32-row strip
  const int nw = wave & 1;         // 0..1 : 64-col strip
  const int lh = lane >> 4;
  const int ll = lane & 15;

  v8f zero = {0.f, 0.f, 0.f, 0.f, 0.f, 0.f, 0.f, 0.f};
  v8f acc[2][4];
#pragma unroll
  for (int i = 0; i < 2; i++)
#pragma unroll
    for (int j = 0; j < 4; j++) acc[i][j] = zero;

  const unsigned asOff = (unsigned)(size_t)(&As[0][0]);
  const unsigned bsOff = (unsigned)(size_t)(&Bs[0][0]);

  for (int k0 = 0; k0 < K; k0 += 32) {
    if (wave == 0) {
      tdm_load_tile_2d(asOff, A  + (size_t)m0 * K + k0, (unsigned long long)K);
      tdm_load_tile_2d(bsOff, Bt + (size_t)n0 * K + k0, (unsigned long long)K);
    }
    __builtin_amdgcn_s_wait_tensorcnt(0);
    __syncthreads();

    Frag af[2];
#pragma unroll
    for (int mt = 0; mt < 2; mt++) {
      int row = mw * 32 + mt * 16 + ll;
      af[mt].q[0] = *(const uint4*)&As[row][8 * lh];
      af[mt].q[1] = *(const uint4*)&As[row][16 + 8 * lh];
    }
#pragma unroll
    for (int nt = 0; nt < 4; nt++) {
      Frag bf;
      int row = nw * 64 + nt * 16 + ll;
      bf.q[0] = *(const uint4*)&Bs[row][8 * lh];
      bf.q[1] = *(const uint4*)&Bs[row][16 + 8 * lh];
#pragma unroll
      for (int mt = 0; mt < 2; mt++)
        acc[mt][nt] = wmma_bf16(af[mt].v, bf.v, acc[mt][nt]);
    }
    __syncthreads();
  }

  // Epilogue.
#pragma unroll
  for (int mt = 0; mt < 2; mt++) {
#pragma unroll
    for (int nt = 0; nt < 4; nt++) {
      int n = n0 + nw * 64 + nt * 16 + ll;
      float bv = bias[n];
#pragma unroll
      for (int r = 0; r < 8; r++) {
        int m = m0 + mw * 32 + mt * 16 + r + 8 * lh;
        float v = acc[mt][nt][r] + bv;
        if (MODE == 0) {
          int sec  = n >> 10;     // 0=q, 1=k, 2=v
          int nn   = n & 1023;
          int head = nn >> 6;
          int d    = nn & 63;
          int b    = m >> 11;     // / TSEQ
          int tt   = m & 2047;
          size_t hb = (size_t)(b * NHEAD + head);
          __bf16 bvv = f2bf(v);
          if (sec == 0)      Qb[(hb * TSEQ + tt) * HD + d] = bvv;
          else if (sec == 1) Kb[(hb * TSEQ + tt) * HD + d] = bvv;
          else               Vt[(hb * HD + d) * TSEQ + tt] = bvv;
        } else {
          out[(size_t)m * N + n] = v;
        }
      }
    }
  }
}

// ---------------------------------------------------------------------------
// Flash attention: grid = (T/64 q-blocks, B*H). 128 threads = 4 waves,
// one wave per 16 q rows. Key blocks of 32 (causal-trimmed loop).
// S = Q K^T via 2 WMMAs per 16x16 tile (hd=64); online softmax in fp32
// (shfl_xor row reduce); P restaged via per-wave LDS; PV via 4 WMMAs.
// ---------------------------------------------------------------------------
__global__ __launch_bounds__(128)
void flash_attn_kernel(const __bf16* __restrict__ Q,
                       const __bf16* __restrict__ Kb,
                       const __bf16* __restrict__ Vt,
                       __bf16* __restrict__ Y /* [B][T][C] bf16 */) {
  __shared__ __bf16 Ps[4][16][40];

  const int bh = blockIdx.y;          // 0..63
  const int b  = bh >> 4;
  const int h  = bh & 15;
  const int wave = threadIdx.x >> 5;
  const int lane = threadIdx.x & 31;
  const int lh = lane >> 4;
  const int ll = lane & 15;
  const int q0 = blockIdx.x * 64 + wave * 16;

  const __bf16* Qh = Q  + (size_t)bh * TSEQ * HD;
  const __bf16* Kh = Kb + (size_t)bh * TSEQ * HD;
  const __bf16* Vh = Vt + (size_t)bh * HD * TSEQ;

  // Q A-fragments for the two hd-chunks (K-dim 0..31 / 32..63). Kept in regs.
  Frag qf[2];
  {
    const __bf16* qrow = Qh + (size_t)(q0 + ll) * HD;
#pragma unroll
    for (int c = 0; c < 2; c++) {
      qf[c].q[0] = *(const uint4*)(qrow + c * 32 + 8 * lh);
      qf[c].q[1] = *(const uint4*)(qrow + c * 32 + 16 + 8 * lh);
    }
  }

  v8f zero = {0.f, 0.f, 0.f, 0.f, 0.f, 0.f, 0.f, 0.f};
  v8f yacc[4];
#pragma unroll
  for (int dt = 0; dt < 4; dt++) yacc[dt] = zero;
  float mrow[8], lrow[8];
#pragma unroll
  for (int r = 0; r < 8; r++) { mrow[r] = -3.0e38f; lrow[r] = 0.f; }

  const float scale = 0.125f;  // 1/sqrt(64)
  const int nkb = (q0 + 16 + 31) >> 5;  // ceil((q0+16)/32) key blocks

  for (int kb = 0; kb < nkb; kb++) {
    const int kbase = kb * 32;

    // K^T B-fragments: lane = key column, contiguous rows of K[T][hd].
    Frag kf[2][2];
#pragma unroll
    for (int nt = 0; nt < 2; nt++) {
      const __bf16* krow = Kh + (size_t)(kbase + nt * 16 + ll) * HD;
#pragma unroll
      for (int c = 0; c < 2; c++) {
        kf[nt][c].q[0] = *(const uint4*)(krow + c * 32 + 8 * lh);
        kf[nt][c].q[1] = *(const uint4*)(krow + c * 32 + 16 + 8 * lh);
      }
    }

    v8f s[2];
#pragma unroll
    for (int nt = 0; nt < 2; nt++) {
      v8f z = zero;
      z = wmma_bf16(qf[0].v, kf[nt][0].v, z);
      z = wmma_bf16(qf[1].v, kf[nt][1].v, z);
      s[nt] = z;
    }

    // Online softmax. Row M = r + 8*lh lives in one 16-lane half -> xor 1..8.
#pragma unroll
    for (int r = 0; r < 8; r++) {
      int qa = q0 + r + 8 * lh;
      float s0 = s[0][r] * scale;
      float s1 = s[1][r] * scale;
      s0 = ((kbase + ll) <= qa) ? s0 : -1.0e30f;
      s1 = ((kbase + 16 + ll) <= qa) ? s1 : -1.0e30f;
      float mx = fmaxf(s0, s1);
      mx = fmaxf(mx, __shfl_xor(mx, 1, 32));
      mx = fmaxf(mx, __shfl_xor(mx, 2, 32));
      mx = fmaxf(mx, __shfl_xor(mx, 4, 32));
      mx = fmaxf(mx, __shfl_xor(mx, 8, 32));
      float mn = fmaxf(mrow[r], mx);
      float p0 = __expf(s0 - mn);
      float p1 = __expf(s1 - mn);
      float alpha = __expf(mrow[r] - mn);
      float rs = p0 + p1;
      rs += __shfl_xor(rs, 1, 32);
      rs += __shfl_xor(rs, 2, 32);
      rs += __shfl_xor(rs, 4, 32);
      rs += __shfl_xor(rs, 8, 32);
      lrow[r] = lrow[r] * alpha + rs;
      mrow[r] = mn;
#pragma unroll
      for (int dt = 0; dt < 4; dt++) yacc[dt][r] *= alpha;
      // Stage P (bf16) for re-fragmenting into WMMA-A layout.
      Ps[wave][r + 8 * lh][ll]      = f2bf(p0);
      Ps[wave][r + 8 * lh][16 + ll] = f2bf(p1);
    }

    // Reload P as A-fragment (same-wave LDS ordering via DScnt).
    Frag pf;
    {
      const __bf16* prow = &Ps[wave][ll][0];
      pf.q[0] = *(const uint4*)(prow + 8 * lh);
      pf.q[1] = *(const uint4*)(prow + 16 + 8 * lh);
    }

    // PV: B-fragments are contiguous rows of V^T[hd][T].
#pragma unroll
    for (int dt = 0; dt < 4; dt++) {
      Frag vf;
      const __bf16* vrow = Vh + (size_t)(dt * 16 + ll) * TSEQ + kbase;
      vf.q[0] = *(const uint4*)(vrow + 8 * lh);
      vf.q[1] = *(const uint4*)(vrow + 16 + 8 * lh);
      yacc[dt] = wmma_bf16(pf.v, vf.v, yacc[dt]);
    }
  }

  // Finalize: divide by l, write bf16 into y[B][T][C] at this head's slice.
#pragma unroll
  for (int dt = 0; dt < 4; dt++) {
#pragma unroll
    for (int r = 0; r < 8; r++) {
      int qa = q0 + r + 8 * lh;
      float v = yacc[dt][r] / lrow[r];
      int col = h * HD + dt * 16 + ll;
      Y[((size_t)b * TSEQ + qa) * CDIM + col] = f2bf(v);
    }
  }
}

// ---------------------------------------------------------------------------
// Host launcher.
// ---------------------------------------------------------------------------
extern "C" void kernel_launch(void* const* d_in, const int* in_sizes, int n_in,
                              void* d_out, int out_size, void* d_ws, size_t ws_size,
                              hipStream_t stream) {
  (void)in_sizes; (void)n_in; (void)out_size; (void)ws_size;

  const float* x      = (const float*)d_in[0];  // [B,T,C]
  const float* W_attn = (const float*)d_in[1];  // [C,3C]
  const float* b_attn = (const float*)d_in[2];  // [3C]
  const float* W_proj = (const float*)d_in[3];  // [C,C]
  const float* b_proj = (const float*)d_in[4];  // [C]
  float* out = (float*)d_out;                   // [B,T,C]

  // Workspace carve-up (bf16 buffers).
  char* w = (char*)d_ws;
  __bf16* xb  = (__bf16*)w; w += (size_t)MTOT * CDIM * 2;         // 16 MB
  __bf16* WaT = (__bf16*)w; w += (size_t)N3C  * CDIM * 2;         //  6 MB
  __bf16* WpT = (__bf16*)w; w += (size_t)CDIM * CDIM * 2;         //  2 MB
  __bf16* Qb  = (__bf16*)w; w += (size_t)MTOT * CDIM * 2;         // 16 MB
  __bf16* Kb  = (__bf16*)w; w += (size_t)MTOT * CDIM * 2;         // 16 MB
  __bf16* Vt  = (__bf16*)w; w += (size_t)MTOT * CDIM * 2;         // 16 MB
  __bf16* yb  = (__bf16*)w; w += (size_t)MTOT * CDIM * 2;         // 16 MB

  // 1) x -> bf16
  {
    int n4 = (MTOT * CDIM) / 4;
    cvt_bf16_kernel<<<dim3(4096), dim3(256), 0, stream>>>(x, xb, n4);
  }
  // 2) W_attn [C][3C] -> WaT [3C][C] bf16 ; W_proj [C][C] -> WpT [C][C] bf16
  transpose_cvt_kernel<<<dim3(N3C / 32, CDIM / 32), dim3(32, 8), 0, stream>>>(
      W_attn, WaT, CDIM, N3C);
  transpose_cvt_kernel<<<dim3(CDIM / 32, CDIM / 32), dim3(32, 8), 0, stream>>>(
      W_proj, WpT, CDIM, CDIM);

  // 3) QKV GEMM with scatter epilogue into Q / K / V^T.
  gemm_bf16_kernel<0><<<dim3(N3C / 128, MTOT / 128), dim3(256), 0, stream>>>(
      xb, WaT, b_attn, Qb, Kb, Vt, nullptr, MTOT, N3C, CDIM);

  // 4) Flash attention -> y (bf16, [B][T][C]).
  flash_attn_kernel<<<dim3(TSEQ / 64, BATCH * NHEAD), dim3(128), 0, stream>>>(
      Qb, Kb, Vt, yb);

  // 5) Output projection -> fp32 d_out.
  gemm_bf16_kernel<1><<<dim3(CDIM / 128, MTOT / 128), dim3(256), 0, stream>>>(
      yb, WpT, b_proj, nullptr, nullptr, nullptr, out, MTOT, CDIM, CDIM);
}
